// SelfAttension_45165876085371
// MI455X (gfx1250) — matrix operand order repaired and checked by reading, not compiled, not run on hardware
//
#include <hip/hip_runtime.h>
#include <hip/hip_bf16.h>

#define NB  2
#define SL  2048
#define NH  16
#define HD  64
#define EMB 1024

typedef __attribute__((ext_vector_type(16))) __bf16 v16bf;
typedef __attribute__((ext_vector_type(8)))  float  v8f;
typedef __attribute__((ext_vector_type(4)))  int    v4i;

// CDNA5 async global->LDS path (ASYNCcnt), guarded so either toolchain compiles.
#if defined(__has_builtin)
#if __has_builtin(__builtin_amdgcn_global_load_async_to_lds_b128) && \
    __has_builtin(__builtin_amdgcn_s_wait_asynccnt)
#define USE_ASYNC 1
#endif
#endif
#ifndef USE_ASYNC
#define USE_ASYNC 0
#endif

// b128 variant takes v4i pointers: global (AS1) source, LDS (AS3) destination
#define ASG(p) ((__attribute__((address_space(1))) v4i*)(p))
#define ASL(p) ((__attribute__((address_space(3))) v4i*)(p))

// 16-bit A-matrix 16x32 layout (ISA 7.12.2): VGPR j holds K pair; half-wave adds +8
__device__ __forceinline__ int a_kidx(int j, int half) {
  return ((j < 4) ? (2 * j) : (16 + 2 * (j - 4))) + 8 * half;
}
// 16-bit B-matrix 32x16 layout: same K order as A (VGPR j holds K pair; half-wave adds +8)
__device__ __forceinline__ int b_kidx(int j, int half) {
  return ((j < 4) ? (2 * j) : (16 + 2 * (j - 4))) + 8 * half;
}

// ---------------------------------------------------------------------------
// Kernel 1: fused QKV projection. X[M=NB*SL*NH rows, 64] * W^T -> bf16.
// blockIdx.y selects (query,Wq)->Qp, (keys,Wk)->Kp, (values,Wv)->Vp.
// (f32 -> bf16 conversion in flight, so staging stays on the VALU path.)
// ---------------------------------------------------------------------------
__global__ void __launch_bounds__(128)
qkv_project_kernel(const float* __restrict__ query, const float* __restrict__ keys,
                   const float* __restrict__ values,
                   const float* __restrict__ Wq, const float* __restrict__ Wk,
                   const float* __restrict__ Wv,
                   __bf16* __restrict__ Qp, __bf16* __restrict__ Kp,
                   __bf16* __restrict__ Vp) {
  const int which = blockIdx.y;
  const float* X = (which == 0) ? query : (which == 1) ? keys : values;
  const float* W = (which == 0) ? Wq : (which == 1) ? Wk : Wv;
  __bf16* Y      = (which == 0) ? Qp : (which == 1) ? Kp : Vp;

  const int mtile = blockIdx.x;            // 16-row tile
  const int tid  = threadIdx.x;
  const int lane = tid & 31, wave = tid >> 5;
  const int half = lane >> 4, ln = lane & 15;

  __shared__ __bf16 At[16][HD + 8];

#pragma unroll
  for (int i = 0; i < 8; ++i) {
    int idx = tid + i * 128;               // 0..1023
    int r = idx >> 6, c = idx & 63;
    At[r][c] = (__bf16)X[(size_t)(mtile * 16 + r) * HD + c];
  }
  __syncthreads();

  v16bf a0, a1;
#pragma unroll
  for (int j = 0; j < 8; ++j) {
    int k0 = a_kidx(j, half);
    a0[2 * j]     = At[ln][k0];
    a0[2 * j + 1] = At[ln][k0 + 1];
    a1[2 * j]     = At[ln][32 + k0];
    a1[2 * j + 1] = At[ln][32 + k0 + 1];
  }

  // y = x @ W^T  =>  B[k][n] = W[n][k]; B column = W row (row-major, contiguous)
  const int e = wave * 16 + ln;
  v16bf b0, b1;
#pragma unroll
  for (int j = 0; j < 8; ++j) {
    int k0 = b_kidx(j, half);
    b0[2 * j]     = (__bf16)W[e * HD + k0];
    b0[2 * j + 1] = (__bf16)W[e * HD + k0 + 1];
    b1[2 * j]     = (__bf16)W[e * HD + 32 + k0];
    b1[2 * j + 1] = (__bf16)W[e * HD + 32 + k0 + 1];
  }

  v8f c = {};
  c = __builtin_amdgcn_wmma_f32_16x16x32_bf16(false, a0, false, b0, (short)0, c, false, false);
  c = __builtin_amdgcn_wmma_f32_16x16x32_bf16(false, a1, false, b1, (short)0, c, false, false);

#pragma unroll
  for (int r = 0; r < 8; ++r) {
    int row = mtile * 16 + r + 8 * half;   // C layout: VGPR r -> M = r (+8 for hi half)
    Y[(size_t)row * HD + wave * 16 + ln] = (__bf16)c[r];
  }
}

// ---------------------------------------------------------------------------
// Kernel 2: flash attention per (n, h). Block = 8 waves = 128 query rows.
// K tile staged via GLOBAL_LOAD_ASYNC_TO_LDS_B128 (ASYNCcnt); V staged
// transposed manually; online softmax over key tiles of 32.
// ---------------------------------------------------------------------------
__global__ void __launch_bounds__(256)
attention_kernel(const __bf16* __restrict__ Qp, const __bf16* __restrict__ Kp,
                 const __bf16* __restrict__ Vp, const int* __restrict__ mask,
                 __bf16* __restrict__ AO) {
  const int n = blockIdx.z, h = blockIdx.y;
  const int qbase = blockIdx.x * 128;
  const int tid  = threadIdx.x;
  const int lane = tid & 31, wave = tid >> 5;
  const int half = lane >> 4, ln = lane & 15;
  const float scale = 0.03125f;            // 1/sqrt(1024)

  __shared__ __bf16 Kt[32][HD];            // K tile row-major, unpadded (async B128)
  __shared__ __bf16 VtT[HD][32 + 4];       // V tile transposed (d-major)
  __shared__ __bf16 Pb[8][16][32 + 4];     // per-wave P relayout buffer

  // Q fragments for this wave's 16 rows (row stride NH*HD = 1024 elems)
  const size_t qrow0 = ((size_t)(n * SL + qbase + wave * 16 + ln) * NH + h) * HD;
  v16bf qa0, qa1;
#pragma unroll
  for (int j = 0; j < 8; ++j) {
    int k0 = a_kidx(j, half);
    qa0[2 * j]     = Qp[qrow0 + k0];
    qa0[2 * j + 1] = Qp[qrow0 + k0 + 1];
    qa1[2 * j]     = Qp[qrow0 + 32 + k0];
    qa1[2 * j + 1] = Qp[qrow0 + 32 + k0 + 1];
  }

  float m[8], l[8];
  v8f o[4] = {};
#pragma unroll
  for (int r = 0; r < 8; ++r) { m[r] = -1e30f; l[r] = 0.0f; }

  for (int kb = 0; kb < SL; kb += 32) {
    __syncthreads();                       // protect tiles from previous iteration

#if USE_ASYNC
    {
      // 32 rows x 128B; 256 lanes each move one 16B chunk, no VGPR data path
      int row = tid >> 3, chunk = tid & 7;
      size_t g = ((size_t)(n * SL + kb + row) * NH + h) * HD + chunk * 8;
      __builtin_amdgcn_global_load_async_to_lds_b128(
          ASG(Kp + g), ASL(&Kt[row][chunk * 8]), 0, 0);
    }
#pragma unroll
    for (int i = 0; i < 8; ++i) {
      int idx = tid + i * 256;             // V: manual transposed staging
      int r = idx >> 6, cc = idx & 63;
      VtT[cc][r] = Vp[((size_t)(n * SL + kb + r) * NH + h) * HD + cc];
    }
#else
#pragma unroll
    for (int i = 0; i < 8; ++i) {
      int idx = tid + i * 256;             // 0..2047
      int r = idx >> 6, cc = idx & 63;
      size_t g = ((size_t)(n * SL + kb + r) * NH + h) * HD + cc;
      Kt[r][cc]  = Kp[g];
      VtT[cc][r] = Vp[g];
    }
#endif
    if (kb + 32 < SL) {                    // gfx1250 global_prefetch_b8 on next tiles
      size_t gn = ((size_t)(n * SL + kb + 32 + (tid >> 3)) * NH + h) * HD;
      __builtin_prefetch(&Kp[gn], 0, 1);
      __builtin_prefetch(&Vp[gn], 0, 1);
    }
#if USE_ASYNC
    __builtin_amdgcn_s_wait_asynccnt(0);   // K tile landed in LDS
#endif
    __syncthreads();

    // scores: two 16x16 tiles over this key block, K-dim = head dim (2 steps)
    v8f s[2];
#pragma unroll
    for (int nt = 0; nt < 2; ++nt) {
      v16bf bk0, bk1;
      int kc = nt * 16 + ln;               // B column = key row
#pragma unroll
      for (int j = 0; j < 8; ++j) {
        int k0 = b_kidx(j, half);
        bk0[2 * j]     = Kt[kc][k0];
        bk0[2 * j + 1] = Kt[kc][k0 + 1];
        bk1[2 * j]     = Kt[kc][32 + k0];
        bk1[2 * j + 1] = Kt[kc][32 + k0 + 1];
      }
      v8f acc = {};
      acc = __builtin_amdgcn_wmma_f32_16x16x32_bf16(false, qa0, false, bk0, (short)0, acc, false, false);
      acc = __builtin_amdgcn_wmma_f32_16x16x32_bf16(false, qa1, false, bk1, (short)0, acc, false, false);
      s[nt] = acc;
    }

    const bool dead0 = (mask[n * SL + kb + ln] == 0);
    const bool dead1 = (mask[n * SL + kb + 16 + ln] == 0);

#pragma unroll
    for (int r = 0; r < 8; ++r) {
      float e0 = dead0 ? -1e20f : s[0][r] * scale;
      float e1 = dead1 ? -1e20f : s[1][r] * scale;
      float mx = fmaxf(e0, e1);            // row lives in a 16-lane half-group
#pragma unroll
      for (int off = 8; off >= 1; off >>= 1) mx = fmaxf(mx, __shfl_xor(mx, off, 16));
      float mnew = fmaxf(m[r], mx);
      float corr = __expf(m[r] - mnew);
      float p0 = __expf(e0 - mnew);
      float p1 = __expf(e1 - mnew);
      float rs = p0 + p1;
#pragma unroll
      for (int off = 8; off >= 1; off >>= 1) rs += __shfl_xor(rs, off, 16);
      l[r] = l[r] * corr + rs;
      m[r] = mnew;
#pragma unroll
      for (int t = 0; t < 4; ++t) o[t][r] *= corr;
      Pb[wave][r + 8 * half][ln]      = (__bf16)p0;   // C-layout -> LDS
      Pb[wave][r + 8 * half][16 + ln] = (__bf16)p1;
    }

    // reload P as A fragment (per-wave private region; wave-internal RAW)
    v16bf pa;
#pragma unroll
    for (int j = 0; j < 8; ++j) {
      int k0 = a_kidx(j, half);
      pa[2 * j]     = Pb[wave][ln][k0];
      pa[2 * j + 1] = Pb[wave][ln][k0 + 1];
    }

    // O += P(16x32) x V(32x64): four 16-col tiles
#pragma unroll
    for (int t = 0; t < 4; ++t) {
      int d = t * 16 + ln;
      v16bf vb;
#pragma unroll
      for (int j = 0; j < 8; ++j) {
        int k0 = b_kidx(j, half);
        vb[2 * j]     = VtT[d][k0];
        vb[2 * j + 1] = VtT[d][k0 + 1];
      }
      o[t] = __builtin_amdgcn_wmma_f32_16x16x32_bf16(false, pa, false, vb, (short)0, o[t], false, false);
    }
  }

#pragma unroll
  for (int r = 0; r < 8; ++r) {
    float inv = 1.0f / l[r];
    int qrow = qbase + wave * 16 + r + 8 * half;
    size_t base = (size_t)(n * SL + qrow) * EMB + h * HD;
#pragma unroll
    for (int t = 0; t < 4; ++t)
      AO[base + t * 16 + ln] = (__bf16)(o[t][r] * inv);
  }
}

// ---------------------------------------------------------------------------
// Kernel 3: output projection  AO[4096x1024](bf16) * Wo^T + bo -> f32 out.
// Block = 4 waves: 16(M) x 64(N) tile; A tile staged via async B128.
// ---------------------------------------------------------------------------
__global__ void __launch_bounds__(128)
out_proj_kernel(const __bf16* __restrict__ AO, const float* __restrict__ Wo,
                const float* __restrict__ bo, float* __restrict__ out) {
  const int mtile = blockIdx.x;
  const int nbase = blockIdx.y * 64;
  const int tid  = threadIdx.x;
  const int lane = tid & 31, wave = tid >> 5;
  const int half = lane >> 4, ln = lane & 15;
  const int e = nbase + wave * 16 + ln;    // output column / Wo row

  __shared__ __bf16 At[16][32];            // unpadded for async B128 staging

  v8f c = {};
  for (int ks = 0; ks < EMB; ks += 32) {
    __syncthreads();
#if USE_ASYNC
    if (tid < 64) {                        // 16 rows x 64B = 64 x 16B chunks
      int r = tid >> 2, chunk = tid & 3;
      __builtin_amdgcn_global_load_async_to_lds_b128(
          ASG(AO + (size_t)(mtile * 16 + r) * EMB + ks + chunk * 8),
          ASL(&At[r][chunk * 8]), 0, 0);
    }
#else
#pragma unroll
    for (int i = 0; i < 4; ++i) {
      int idx = tid + i * 128;             // 0..511
      int r = idx >> 5, cc = idx & 31;
      At[r][cc] = AO[(size_t)(mtile * 16 + r) * EMB + ks + cc];
    }
#endif
    if (ks + 32 < EMB)
      __builtin_prefetch(&AO[(size_t)(mtile * 16 + (tid >> 3)) * EMB + ks + 32], 0, 1);
#if USE_ASYNC
    __builtin_amdgcn_s_wait_asynccnt(0);
#endif
    __syncthreads();

    v16bf a, b;
#pragma unroll
    for (int j = 0; j < 8; ++j) {
      int ka = a_kidx(j, half);
      a[2 * j]     = At[ln][ka];
      a[2 * j + 1] = At[ln][ka + 1];
      int kg = ks + b_kidx(j, half);       // B[k][e] = Wo[e][k], row-major
      b[2 * j]     = (__bf16)Wo[(size_t)e * EMB + kg];
      b[2 * j + 1] = (__bf16)Wo[(size_t)e * EMB + kg + 1];
    }
    c = __builtin_amdgcn_wmma_f32_16x16x32_bf16(false, a, false, b, (short)0, c, false, false);
  }

#pragma unroll
  for (int r = 0; r < 8; ++r) {
    int row = mtile * 16 + r + 8 * half;
    out[(size_t)row * EMB + e] = c[r] + bo[e];
  }
}

// ---------------------------------------------------------------------------
extern "C" void kernel_launch(void* const* d_in, const int* in_sizes, int n_in,
                              void* d_out, int out_size, void* d_ws, size_t ws_size,
                              hipStream_t stream) {
  (void)in_sizes; (void)n_in; (void)out_size; (void)ws_size;
  const float* values = (const float*)d_in[0];
  const float* keys   = (const float*)d_in[1];
  const float* query  = (const float*)d_in[2];
  const int*   mask   = (const int*)d_in[3];
  const float* Wv     = (const float*)d_in[4];
  const float* Wk     = (const float*)d_in[5];
  const float* Wq     = (const float*)d_in[6];
  const float* Wo     = (const float*)d_in[7];
  const float* bo     = (const float*)d_in[8];
  float* out = (float*)d_out;

  const size_t nelem = (size_t)NB * SL * EMB;        // 4M bf16 per tensor
  char* ws = (char*)d_ws;
  __bf16* Qp = (__bf16*)(ws + 0 * nelem * 2);
  __bf16* Kp = (__bf16*)(ws + 1 * nelem * 2);
  __bf16* Vp = (__bf16*)(ws + 2 * nelem * 2);
  __bf16* AO = (__bf16*)(ws + 3 * nelem * 2);

  dim3 g1((NB * SL * NH) / 16, 3);
  qkv_project_kernel<<<g1, 128, 0, stream>>>(query, keys, values, Wq, Wk, Wv, Qp, Kp, Vp);

  dim3 g2(SL / 128, NH, NB);
  attention_kernel<<<g2, 256, 0, stream>>>(Qp, Kp, Vp, mask, AO);

  dim3 g3((NB * SL) / 16, EMB / 64);
  out_proj_kernel<<<g3, 128, 0, stream>>>(AO, Wo, bo, out);
}
